// LoRACrossAttnProcessor_10986526343841
// MI455X (gfx1250) — matrix operand order repaired and checked by reading, not compiled, not run on hardware
//
#include <hip/hip_runtime.h>

// ---------------------------------------------------------------------------
// MI455X (gfx1250) fused LoRA self-attention.
//   W_eff = W + up@down  (LoRA folded into the weight, rank 4)
//   QKV GEMMs + LDS-staged flash attention + output GEMM,
//   all matrix math via v_wmma_f32_16x16x32_f16 (f32 accumulate).
//   Attention K/V tiles staged with GLOBAL_LOAD_ASYNC_TO_LDS_B128
//   (ASYNCcnt-tracked direct global->LDS DMA).
// ---------------------------------------------------------------------------

typedef _Float16 v16h __attribute__((ext_vector_type(16)));
typedef _Float16 v8h  __attribute__((ext_vector_type(8)));
typedef float    v8f  __attribute__((ext_vector_type(8)));
typedef int      v4i  __attribute__((vector_size(16)));

#define D_MODEL 640
#define N_HEADS 8
#define HEAD_DIM 80
#define SEQ 2048
#define BATCH 4
#define M_TOTAL (BATCH * SEQ) /* 8192 */

#define CAT16(lo, hi) \
  __builtin_shufflevector((lo), (hi), 0,1,2,3,4,5,6,7,8,9,10,11,12,13,14,15)
#define WMMA_F16(a, b, c) \
  __builtin_amdgcn_wmma_f32_16x16x32_f16(false, (a), false, (b), (short)0, (c), false, false)

#if __has_builtin(__builtin_amdgcn_global_load_async_to_lds_b128) && \
    __has_builtin(__builtin_amdgcn_s_wait_asynccnt)
#define HAVE_ASYNC_LDS 1
typedef __attribute__((address_space(1))) v4i* as1_v4i_ptr;
typedef __attribute__((address_space(3))) v4i* as3_v4i_ptr;
#define ASYNC_CP128(dst, src)                                   \
  __builtin_amdgcn_global_load_async_to_lds_b128(               \
      (as1_v4i_ptr)(src), (as3_v4i_ptr)(dst), 0, 0)
#endif

// ---------------- prep: W_eff = W + up@down, cast to f16 -------------------
__global__ void build_weff(const float* __restrict__ W,
                           const float* __restrict__ down,  // [4, 640]
                           const float* __restrict__ up,    // [640, 4]
                           _Float16* __restrict__ Wh) {
  int idx = blockIdx.x * blockDim.x + threadIdx.x;
  if (idx >= D_MODEL * D_MODEL) return;
  int n = idx / D_MODEL, k = idx % D_MODEL;
  float acc = W[idx];
#pragma unroll
  for (int r = 0; r < 4; ++r) acc += up[n * 4 + r] * down[r * D_MODEL + k];
  Wh[idx] = (_Float16)acc;
}

__global__ void cast_f32_f16(const float* __restrict__ x,
                             _Float16* __restrict__ y, int n) {
  int idx = blockIdx.x * blockDim.x + threadIdx.x;
  if (idx < n) y[idx] = (_Float16)x[idx];
}

// ---------------- GEMM: Y[M,640] = A[M,640] @ W[640,640]^T -----------------
// mode 0: store f16 row-major [M,640]
// mode 1: store f16 transposed per head: Vt[b][h][e][s]
// mode 2: store f32 + bias to out (final projection)
__global__ __launch_bounds__(64)
void gemm_tn(const _Float16* __restrict__ A,
             const _Float16* __restrict__ W,
             const float* __restrict__ bias,
             _Float16* __restrict__ oh,
             float* __restrict__ of,
             int mode) {
  const int lane = threadIdx.x & 31;
  const int wave = threadIdx.x >> 5;
  const int g = lane >> 4;  // lane group (0: lanes 0-15, 1: lanes 16-31)
  const int nn = lane & 15;
  const int m0 = blockIdx.x * 16;
  const int n0 = blockIdx.y * 128 + wave * 64;

  v8f acc[4] = {v8f{}, v8f{}, v8f{}, v8f{}};
  const _Float16* arow = A + (size_t)(m0 + nn) * D_MODEL;
  const _Float16* brow[4];
#pragma unroll
  for (int t = 0; t < 4; ++t)
    brow[t] = W + (size_t)(n0 + t * 16 + nn) * D_MODEL;

#pragma unroll
  for (int k0 = 0; k0 < D_MODEL; k0 += 32) {
    // A fragment (16x32 f16): lane m=nn; halves 0..7 -> K=g*8+i, 8..15 -> 16+g*8+i
    v8h alo = *(const v8h*)(arow + k0 + g * 8);
    v8h ahi = *(const v8h*)(arow + k0 + 16 + g * 8);
    v16h af = CAT16(alo, ahi);
#pragma unroll
    for (int t = 0; t < 4; ++t) {
      // B fragment (32x16): B[k][n] = W[n][k] -> row-contiguous halves
      v8h blo = *(const v8h*)(brow[t] + k0 + g * 8);
      v8h bhi = *(const v8h*)(brow[t] + k0 + 16 + g * 8);
      acc[t] = WMMA_F16(af, CAT16(blo, bhi), acc[t]);
    }
  }

#pragma unroll
  for (int t = 0; t < 4; ++t) {
    const int d = n0 + t * 16 + nn;  // output column (lane-fixed)
    if (mode == 0) {
#pragma unroll
      for (int r = 0; r < 8; ++r) {
        int m = m0 + g * 8 + r;  // C layout: lanes 0-15 rows 0..7, lanes 16-31 rows 8..15
        oh[(size_t)m * D_MODEL + d] = (_Float16)acc[t][r];
      }
    } else if (mode == 1) {
      // V transposed: Vt[((b*H + h)*80 + e)*2048 + s]; rows are consecutive s
      int e = d % HEAD_DIM, hh = d / HEAD_DIM;
      int m = m0 + g * 8;
      int b = m >> 11, sl = m & (SEQ - 1);
      v8h vals;
#pragma unroll
      for (int r = 0; r < 8; ++r) vals[r] = (_Float16)acc[t][r];
      *(v8h*)(oh + ((size_t)(b * N_HEADS + hh) * HEAD_DIM + e) * SEQ + sl) = vals;
    } else {
      float bv = bias[d];
#pragma unroll
      for (int r = 0; r < 8; ++r) {
        int m = m0 + g * 8 + r;
        of[(size_t)m * D_MODEL + d] = acc[t][r] + bv;
      }
    }
  }
}

// ---------------- flash attention, LDS-staged K/V tiles --------------------
// Block = 4 waves = 64 queries of one (b,h); K/V tiles (32 keys) are staged
// in LDS once per block (shared by the 4 waves) with double buffering.
#define KT_STRIDE 88  /* padded from 80: bank-conflict-free ds_load_b128 */
#define VT_STRIDE 40  /* padded from 32 */
#define NIT (SEQ / 32)

__global__ __launch_bounds__(128)
void attn_fused(const _Float16* __restrict__ Qh,  // [B*S, 640]
                const _Float16* __restrict__ Kh,  // [B*S, 640]
                const _Float16* __restrict__ Vt,  // [B,H,80,2048]
                _Float16* __restrict__ Oh) {      // [B*S, 640]
  __shared__ _Float16 KtLds[2][32 * KT_STRIDE];
  __shared__ _Float16 VtLds[2][HEAD_DIM * VT_STRIDE];
  __shared__ _Float16 Pld[4][16 * 32];

  const int tid = threadIdx.x;
  const int lane = tid & 31, wave = tid >> 5;
  const int g = lane >> 4, nn = lane & 15;
  const int q0 = blockIdx.x * 64 + wave * 16;
  const int h = blockIdx.y;
  const int b = blockIdx.z;
  const float sm_scale = 0.11180339887498948f;  // 1/sqrt(80)

  const _Float16* kbase = Kh + (size_t)(b * SEQ) * D_MODEL + h * HEAD_DIM;
  const _Float16* vbase = Vt + (size_t)((b * N_HEADS + h) * HEAD_DIM) * SEQ;

  // Q fragments: hd=80 = 32 + 32 + 16 (tail chunk: upper halves are dims 80..95 -> 0)
  v16h qf[3];
  {
    const _Float16* qr = Qh + (size_t)(b * SEQ + q0 + nn) * D_MODEL + h * HEAD_DIM;
#pragma unroll
    for (int c = 0; c < 3; ++c) {
      v8h lo = *(const v8h*)(qr + c * 32 + g * 8);
      v8h hi = {};
      if (c < 2) hi = *(const v8h*)(qr + c * 32 + 16 + g * 8);
      qf[c] = CAT16(lo, hi);
    }
  }

  v8f ofr[5] = {v8f{}, v8f{}, v8f{}, v8f{}, v8f{}};
  float mst[8], lst[8];
#pragma unroll
  for (int r = 0; r < 8; ++r) { mst[r] = -1e30f; lst[r] = 0.0f; }

  // ---- tile staging: K = 32 keys x 10 B128 chunks, V = 80 dims x 4 chunks ----
#ifdef HAVE_ASYNC_LDS
  // Direct global->LDS DMA, no VGPR round trip; tracked by ASYNCcnt.
#define STAGE_TILES(buf, kt_)                                                 \
  do {                                                                        \
    _Pragma("unroll") for (int c = 0; c < 3; ++c) {                           \
      int idx = tid + c * 128;                                                \
      if (idx < 320) {                                                        \
        int key = idx / 10, off = (idx % 10) * 8;                             \
        ASYNC_CP128(&KtLds[buf][key * KT_STRIDE + off],                       \
                    kbase + (size_t)((kt_) + key) * D_MODEL + off);           \
      }                                                                       \
    }                                                                         \
    _Pragma("unroll") for (int c = 0; c < 3; ++c) {                           \
      int idx = tid + c * 128;                                                \
      if (idx < 320) {                                                        \
        int e = idx / 4, off = (idx % 4) * 8;                                 \
        ASYNC_CP128(&VtLds[buf][e * VT_STRIDE + off],                         \
                    vbase + (size_t)e * SEQ + (kt_) + off);                   \
      }                                                                       \
    }                                                                         \
  } while (0)

  STAGE_TILES(0, 0);
  __builtin_amdgcn_s_wait_asynccnt(0);
#else
#define STAGE_TILES(buf, kt_)                                                 \
  do {                                                                        \
    _Pragma("unroll") for (int c = 0; c < 3; ++c) {                           \
      int idx = tid + c * 128;                                                \
      if (idx < 320) {                                                        \
        int key = idx / 10, off = (idx % 10) * 8;                             \
        *(v8h*)&KtLds[buf][key * KT_STRIDE + off] =                           \
            *(const v8h*)(kbase + (size_t)((kt_) + key) * D_MODEL + off);     \
      }                                                                       \
    }                                                                         \
    _Pragma("unroll") for (int c = 0; c < 3; ++c) {                           \
      int idx = tid + c * 128;                                                \
      if (idx < 320) {                                                        \
        int e = idx / 4, off = (idx % 4) * 8;                                 \
        *(v8h*)&VtLds[buf][e * VT_STRIDE + off] =                             \
            *(const v8h*)(vbase + (size_t)e * SEQ + (kt_) + off);             \
      }                                                                       \
    }                                                                         \
  } while (0)

  STAGE_TILES(0, 0);
#endif
  __syncthreads();

  for (int it = 0; it < NIT; ++it) {
    const int kt = it * 32;
    const int p = it & 1;
    const bool more = (it + 1) < NIT;

    // ---- start staging tile it+1 into the other buffer ----
    // (async path: DMA proceeds during compute; fallback: loads land in
    //  registers here and are committed to LDS after compute)
#ifdef HAVE_ASYNC_LDS
    if (more) STAGE_TILES(1 - p, kt + 32);
#else
    v8h kpre[3], vpre[3];
    if (more) {
      const int kt2 = kt + 32;
#pragma unroll
      for (int c = 0; c < 3; ++c) {
        int idx = tid + c * 128;
        if (idx < 320) {
          int key = idx / 10, off = (idx % 10) * 8;
          kpre[c] = *(const v8h*)(kbase + (size_t)(kt2 + key) * D_MODEL + off);
        }
      }
#pragma unroll
      for (int c = 0; c < 3; ++c) {
        int idx = tid + c * 128;
        if (idx < 320) {
          int e = idx / 4, off = (idx % 4) * 8;
          vpre[c] = *(const v8h*)(vbase + (size_t)e * SEQ + kt2 + off);
        }
      }
    }
#endif

    // ---- scores: two 16x16 tiles from LDS ----
    v8f s01[2];
#pragma unroll
    for (int half = 0; half < 2; ++half) {
      const _Float16* kr = &KtLds[p][(half * 16 + nn) * KT_STRIDE];
      v8f s = {};
#pragma unroll
      for (int c = 0; c < 3; ++c) {
        v16h bf;
        if (c < 2) {
          v8h lo = *(const v8h*)(kr + c * 32 + g * 8);
          v8h hi = *(const v8h*)(kr + c * 32 + 16 + g * 8);
          bf = CAT16(lo, hi);
        } else {
          // dims 64..79: lane-group g holds dims 64+8g..71+8g in the low half
          v8h lo = *(const v8h*)(kr + 64 + g * 8), hi = {};
          bf = CAT16(lo, hi);
        }
        s = WMMA_F16(qf[c], bf, s);
      }
      s01[half] = s;
    }
    v8f s0 = s01[0], s1 = s01[1];

    // ---- online softmax (row = 8*g + r, spread over 16 lanes) ----
    float mnew[8], alpha[8];
#pragma unroll
    for (int r = 0; r < 8; ++r) {
      s0[r] *= sm_scale;
      s1[r] *= sm_scale;
      float v = fmaxf(s0[r], s1[r]);
#pragma unroll
      for (int off = 1; off < 16; off <<= 1) v = fmaxf(v, __shfl_xor(v, off, 32));
      mnew[r] = fmaxf(mst[r], v);
      alpha[r] = __expf(mst[r] - mnew[r]);
      mst[r] = mnew[r];
    }
#pragma unroll
    for (int r = 0; r < 8; ++r) {
      float p0 = __expf(s0[r] - mnew[r]);
      float p1 = __expf(s1[r] - mnew[r]);
      s0[r] = p0; s1[r] = p1;
      float rs = p0 + p1;
#pragma unroll
      for (int off = 1; off < 16; off <<= 1) rs += __shfl_xor(rs, off, 32);
      lst[r] = lst[r] * alpha[r] + rs;
    }
#pragma unroll
    for (int r = 0; r < 8; ++r) {
#pragma unroll
      for (int t = 0; t < 5; ++t) ofr[t][r] *= alpha[r];
    }

    // ---- relayout P: C-fragment(f32) -> LDS -> A-fragment(f16) ----
#pragma unroll
    for (int r = 0; r < 8; ++r) {
      Pld[wave][(g * 8 + r) * 32 + nn] = (_Float16)s0[r];
      Pld[wave][(g * 8 + r) * 32 + 16 + nn] = (_Float16)s1[r];
    }
    v8h plo = *(const v8h*)(&Pld[wave][nn * 32 + g * 8]);
    v8h phi = *(const v8h*)(&Pld[wave][nn * 32 + 16 + g * 8]);
    v16h pf = CAT16(plo, phi);

    // ---- O += P(16x32) @ V(32x80): 5 WMMA, B fragments from LDS ----
#pragma unroll
    for (int t = 0; t < 5; ++t) {
      const _Float16* vr = &VtLds[p][(t * 16 + nn) * VT_STRIDE + g * 8];
      v8h lo = *(const v8h*)(vr);
      v8h hi = *(const v8h*)(vr + 16);
      v16h vf = CAT16(lo, hi);
      ofr[t] = WMMA_F16(pf, vf, ofr[t]);
    }

    // ---- make tile it+1 visible to all waves ----
#ifdef HAVE_ASYNC_LDS
    if (more) __builtin_amdgcn_s_wait_asynccnt(0);
#else
    if (more) {
#pragma unroll
      for (int c = 0; c < 3; ++c) {
        int idx = tid + c * 128;
        if (idx < 320) {
          int key = idx / 10, off = (idx % 10) * 8;
          *(v8h*)&KtLds[1 - p][key * KT_STRIDE + off] = kpre[c];
        }
      }
#pragma unroll
      for (int c = 0; c < 3; ++c) {
        int idx = tid + c * 128;
        if (idx < 320) {
          int e = idx / 4, off = (idx % 4) * 8;
          *(v8h*)&VtLds[1 - p][e * VT_STRIDE + off] = vpre[c];
        }
      }
    }
#endif
    __syncthreads();
  }

  // ---- normalize and store f16 [B*S, 640] ----
  _Float16* orow = Oh + (size_t)(b * SEQ) * D_MODEL + h * HEAD_DIM;
#pragma unroll
  for (int r = 0; r < 8; ++r) {
    float inv = 1.0f / lst[r];
#pragma unroll
    for (int t = 0; t < 5; ++t) {
      orow[(size_t)(q0 + g * 8 + r) * D_MODEL + t * 16 + nn] =
          (_Float16)(ofr[t][r] * inv);
    }
  }
}

// ---------------------------------------------------------------------------
extern "C" void kernel_launch(void* const* d_in, const int* in_sizes, int n_in,
                              void* d_out, int out_size, void* d_ws, size_t ws_size,
                              hipStream_t stream) {
  const float* x  = (const float*)d_in[0];
  const float* wq = (const float*)d_in[1];
  const float* wk = (const float*)d_in[2];
  const float* wv = (const float*)d_in[3];
  const float* wo = (const float*)d_in[4];
  const float* bo = (const float*)d_in[5];
  const float* qd = (const float*)d_in[6];
  const float* qu = (const float*)d_in[7];
  const float* kd = (const float*)d_in[8];
  const float* ku = (const float*)d_in[9];
  const float* vd = (const float*)d_in[10];
  const float* vu = (const float*)d_in[11];
  const float* od = (const float*)d_in[12];
  const float* ou = (const float*)d_in[13];
  float* out = (float*)d_out;

  // workspace layout (f16): Xh, Qh, Kh, Vt, AttnH (each M*640) + 4 W_eff
  const size_t NX = (size_t)M_TOTAL * D_MODEL;  // 5,242,880 elems
  _Float16* Xh = (_Float16*)d_ws;
  _Float16* Qh = Xh + NX;
  _Float16* Kh = Qh + NX;
  _Float16* Vtb = Kh + NX;
  _Float16* Ah = Vtb + NX;
  _Float16* Wqh = Ah + NX;
  _Float16* Wkh = Wqh + (size_t)D_MODEL * D_MODEL;
  _Float16* Wvh = Wkh + (size_t)D_MODEL * D_MODEL;
  _Float16* Woh = Wvh + (size_t)D_MODEL * D_MODEL;

  const int wblocks = (D_MODEL * D_MODEL + 255) / 256;
  build_weff<<<wblocks, 256, 0, stream>>>(wq, qd, qu, Wqh);
  build_weff<<<wblocks, 256, 0, stream>>>(wk, kd, ku, Wkh);
  build_weff<<<wblocks, 256, 0, stream>>>(wv, vd, vu, Wvh);
  build_weff<<<wblocks, 256, 0, stream>>>(wo, od, ou, Woh);
  cast_f32_f16<<<(int)((NX + 255) / 256), 256, 0, stream>>>(x, Xh, (int)NX);

  dim3 ggrid(M_TOTAL / 16, D_MODEL / 128);  // (512, 5), 2 waves/block
  gemm_tn<<<ggrid, 64, 0, stream>>>(Xh, Wqh, nullptr, Qh, nullptr, 0);
  gemm_tn<<<ggrid, 64, 0, stream>>>(Xh, Wkh, nullptr, Kh, nullptr, 0);
  gemm_tn<<<ggrid, 64, 0, stream>>>(Xh, Wvh, nullptr, Vtb, nullptr, 1);

  dim3 agrid(SEQ / 64, N_HEADS, BATCH);  // 1024 blocks x 4 waves
  attn_fused<<<agrid, 128, 0, stream>>>(Qh, Kh, Vtb, Ah);

  gemm_tn<<<ggrid, 64, 0, stream>>>(Ah, Woh, bo, nullptr, out, 2);
}
